// LlamaDecoderLayer_70738111365900
// MI455X (gfx1250) — matrix operand order repaired and checked by reading, not compiled, not run on hardware
//
#include <hip/hip_runtime.h>
#include <hip/hip_fp16.h>
#include <hip/hip_bf16.h>

// ---------------- model dims (from reference) ----------------
#define BATCH   4
#define SEQ     1024
#define HDIM    2048
#define NHEADS  16
#define HEADD   128          // HD
#define ROPED   64           // RD
#define LATD    512          // LAT
#define IDIM    1024         // I
#define NREXP   7            // routed experts (NE - NSH)
#define TOKENS  (BATCH * SEQ)
#define QSCALE  0.08838834764831845f   // 1/sqrt(128)
#define EPSV    1e-5f
#define INV_SCALING 0.125f   // 1/8

typedef __attribute__((ext_vector_type(16))) _Float16 v16h;
typedef __attribute__((ext_vector_type(8)))  float    v8f;
typedef __attribute__((ext_vector_type(4)))  int      v4i;

// ---------------- CDNA5 async global->LDS copy (b128) ----------------
#if defined(__has_builtin)
#if __has_builtin(__builtin_amdgcn_global_load_async_to_lds_b128)
#define HAVE_ASYNC_LDS 1
#endif
#if __has_builtin(__builtin_amdgcn_s_wait_asynccnt)
#define HAVE_WAIT_ASYNC 1
#endif
#endif

__device__ __forceinline__ void wait_async0() {
#if defined(HAVE_ASYNC_LDS)
#if defined(HAVE_WAIT_ASYNC)
  __builtin_amdgcn_s_wait_asynccnt(0);
#else
  asm volatile("s_wait_asynccnt 0x0" ::: "memory");
#endif
#endif
}

// copy 16 bytes from global to LDS (per-lane). Async on gfx1250 if available.
__device__ __forceinline__ void copy16_g2l(const _Float16* gsrc, _Float16* ldst) {
#if defined(HAVE_ASYNC_LDS)
  __attribute__((address_space(1))) void* g1 =
      (__attribute__((address_space(1))) void*)((void*)gsrc);
  __attribute__((address_space(3))) void* l3 =
      (__attribute__((address_space(3))) void*)((void*)ldst);
  __builtin_amdgcn_global_load_async_to_lds_b128(
      (__attribute__((address_space(1))) v4i*)g1,
      (__attribute__((address_space(3))) v4i*)l3, 0, 0);
#else
  *(uint4*)ldst = *(const uint4*)gsrc;
#endif
}

// =====================================================================
// elementwise: f32 -> f16
// =====================================================================
__global__ __launch_bounds__(256) void cvt_f32_f16(const float* __restrict__ src,
                                                   _Float16* __restrict__ dst, long n) {
  long i = (long)blockIdx.x * blockDim.x + threadIdx.x;
  long stride = (long)gridDim.x * blockDim.x;
  for (; i < n; i += stride) dst[i] = (_Float16)src[i];
}

// =====================================================================
// rmsnorm: per-token row of H, f32 in, f16 out
// =====================================================================
__global__ __launch_bounds__(256) void rmsnorm_kernel(const float* __restrict__ x,
                                                      const float* __restrict__ w,
                                                      _Float16* __restrict__ out) {
  int tk = blockIdx.x;
  const float* xr = x + (long)tk * HDIM;
  __shared__ float red[256];
  float s = 0.f;
  for (int i = threadIdx.x; i < HDIM; i += 256) { float v = xr[i]; s += v * v; }
  red[threadIdx.x] = s; __syncthreads();
  for (int st = 128; st > 0; st >>= 1) {
    if (threadIdx.x < st) red[threadIdx.x] += red[threadIdx.x + st];
    __syncthreads();
  }
  float r = rsqrtf(red[0] * (1.0f / HDIM) + EPSV);
  for (int i = threadIdx.x; i < HDIM; i += 256)
    out[(long)tk * HDIM + i] = (_Float16)(w[i] * r * xr[i]);
}

// =====================================================================
// residual add + rmsnorm:  h = x + p ; hn = rmsnorm(h, w)  (hn in f16)
// =====================================================================
__global__ __launch_bounds__(256) void add_rmsnorm_kernel(const float* __restrict__ x,
                                                          const float* __restrict__ p,
                                                          const float* __restrict__ w,
                                                          float* __restrict__ h,
                                                          _Float16* __restrict__ hn) {
  int tk = blockIdx.x;
  long base = (long)tk * HDIM;
  __shared__ float red[256];
  float s = 0.f;
  for (int i = threadIdx.x; i < HDIM; i += 256) {
    float v = x[base + i] + p[base + i];
    h[base + i] = v;
    s += v * v;
  }
  red[threadIdx.x] = s; __syncthreads();
  for (int st = 128; st > 0; st >>= 1) {
    if (threadIdx.x < st) red[threadIdx.x] += red[threadIdx.x + st];
    __syncthreads();
  }
  float r = rsqrtf(red[0] * (1.0f / HDIM) + EPSV);
  for (int i = threadIdx.x; i < HDIM; i += 256)
    hn[base + i] = (_Float16)(w[i] * r * h[base + i]);
}

// =====================================================================
// Generic f16 WMMA GEMM:  C[M,N] = A[M,K] @ B[K,N], f32 accumulate.
// Block = 256 threads = 8 waves (2x4), block tile 64(M) x 128(N),
// each wave computes 32x32 via 4 accumulators. K multiple of 32,
// M multiple of 64, N multiple of 128. Row-major with explicit ld*.
// blockIdx.z batches with element strides sA/sB/sC.
// Double-buffered LDS tiles filled by GLOBAL_LOAD_ASYNC_TO_LDS_B128
// (tracked by ASYNCcnt) so the next tile's fill overlaps the WMMAs.
// =====================================================================
template <bool OUT_HALF>
__global__ __launch_bounds__(256) void gemm_f16(const _Float16* __restrict__ A,
                                                const _Float16* __restrict__ B,
                                                void* __restrict__ Cv,
                                                int M, int N, int K,
                                                int lda, int ldb, int ldc,
                                                long sA, long sB, long sC) {
  A += (long)blockIdx.z * sA;
  B += (long)blockIdx.z * sB;
  float*    Cf = (float*)Cv    + (long)blockIdx.z * sC;
  _Float16* Ch = (_Float16*)Cv + (long)blockIdx.z * sC;

  __shared__ __align__(16) _Float16 As[2][64 * 40];   // 64 x 32 (+8 pad/row)
  __shared__ __align__(16) _Float16 Bs[2][32 * 136];  // 32 x 128 (+8 pad/row)

  const int tid = threadIdx.x;
  const int wave = tid >> 5, lane = tid & 31;
  const int laneRow = lane & 15, laneHi = lane >> 4;
  const int wm = wave >> 2, wn = wave & 3;       // wave grid 2(M) x 4(N)
  const int m0 = blockIdx.y * 64;
  const int n0 = blockIdx.x * 128;

  // tile-load indexing (8-half = 16B vectors)
  const int arow = tid >> 2;          // 0..63
  const int acol = (tid & 3) * 8;     // 0,8,16,24
  const int brow = tid >> 4;          // 0..15
  const int bcol = (tid & 15) * 8;    // 0..120

  auto issue_tile = [&](int buf, int k0) {
    copy16_g2l(&A[(long)(m0 + arow) * lda + k0 + acol], &As[buf][arow * 40 + acol]);
    copy16_g2l(&B[(long)(k0 + brow) * ldb + n0 + bcol], &Bs[buf][brow * 136 + bcol]);
    copy16_g2l(&B[(long)(k0 + brow + 16) * ldb + n0 + bcol],
               &Bs[buf][(brow + 16) * 136 + bcol]);
  };

  v8f c00 = {}, c01 = {}, c10 = {}, c11 = {};

  issue_tile(0, 0);
  for (int k0 = 0; k0 < K; k0 += 32) {
    const int buf = (k0 >> 5) & 1;
    wait_async0();          // this wave's fills of As/Bs[buf] have landed
    __syncthreads();        // ... and every other wave's too
    if (k0 + 32 < K) issue_tile(buf ^ 1, k0 + 32);   // overlap next fill

    v16h a0, a1, b0, b1;
    // A 16x32 f16 layout: lane L(0-15)=row M=L; VGPR i holds
    // K = (i>>2)*16 + laneHi*8 + (i&3)*2 (+1 in high half)
#pragma unroll
    for (int i = 0; i < 8; ++i) {
      int kk = ((i >> 2) << 4) + (laneHi << 3) + ((i & 3) << 1);
      a0[2 * i]     = As[buf][(wm * 32 + laneRow) * 40 + kk];
      a0[2 * i + 1] = As[buf][(wm * 32 + laneRow) * 40 + kk + 1];
      a1[2 * i]     = As[buf][(wm * 32 + 16 + laneRow) * 40 + kk];
      a1[2 * i + 1] = As[buf][(wm * 32 + 16 + laneRow) * 40 + kk + 1];
    }
    // B 32x16 f16 layout: VGPR j holds K = (j>>2)*16 + laneHi*8 + (j&3)*2 (+1), N = laneRow
#pragma unroll
    for (int j = 0; j < 8; ++j) {
      int kk = ((j >> 2) << 4) + (laneHi << 3) + ((j & 3) << 1);
      b0[2 * j]     = Bs[buf][kk * 136 + wn * 32 + laneRow];
      b0[2 * j + 1] = Bs[buf][(kk + 1) * 136 + wn * 32 + laneRow];
      b1[2 * j]     = Bs[buf][kk * 136 + wn * 32 + 16 + laneRow];
      b1[2 * j + 1] = Bs[buf][(kk + 1) * 136 + wn * 32 + 16 + laneRow];
    }
    c00 = __builtin_amdgcn_wmma_f32_16x16x32_f16(false, a0, false, b0, (short)0, c00, false, false);
    c01 = __builtin_amdgcn_wmma_f32_16x16x32_f16(false, a0, false, b1, (short)0, c01, false, false);
    c10 = __builtin_amdgcn_wmma_f32_16x16x32_f16(false, a1, false, b0, (short)0, c10, false, false);
    c11 = __builtin_amdgcn_wmma_f32_16x16x32_f16(false, a1, false, b1, (short)0, c11, false, false);
    __syncthreads();        // everyone done reading buf before it is refilled
  }

  // C 16x16 f32 layout: VGPR r -> M = laneHi*8 + r, N = laneRow
  const int mb = m0 + wm * 32;
  const int nb = n0 + wn * 32;
#pragma unroll
  for (int r = 0; r < 8; ++r) {
    int mm = mb + (laneHi << 3) + r;
    int nn = nb + laneRow;
    if constexpr (OUT_HALF) {
      Ch[(long)mm * ldc + nn]             = (_Float16)c00[r];
      Ch[(long)mm * ldc + nn + 16]        = (_Float16)c01[r];
      Ch[(long)(mm + 16) * ldc + nn]      = (_Float16)c10[r];
      Ch[(long)(mm + 16) * ldc + nn + 16] = (_Float16)c11[r];
    } else {
      Cf[(long)mm * ldc + nn]             = c00[r];
      Cf[(long)mm * ldc + nn + 16]        = c01[r];
      Cf[(long)(mm + 16) * ldc + nn]      = c10[r];
      Cf[(long)(mm + 16) * ldc + nn + 16] = c11[r];
    }
  }
}

// =====================================================================
// RoPE + concat + q-scale packing.
// Inputs (f16): kp/qp [tok, NH*64], kr/qr [tok, NH*64]
// Outputs (f16): kout/qout [tok, NH*128]  (first 64 = compressed part,
//   last 64 = roped part); q gets *QSCALE folded in.
// =====================================================================
__global__ __launch_bounds__(256) void rope_pack_kernel(const _Float16* __restrict__ kp,
                                                        const _Float16* __restrict__ qp,
                                                        const _Float16* __restrict__ kr,
                                                        const _Float16* __restrict__ qr,
                                                        _Float16* __restrict__ kout,
                                                        _Float16* __restrict__ qout) {
  int tk = blockIdx.x;
  int s = tk & (SEQ - 1);
  for (int it = threadIdx.x; it < NHEADS * 32; it += 256) {
    int h = it >> 5;
    int d2 = it & 31;                 // 0..31 -> pairs (d2, d2+32)
    float inv = powf(10000.f, -(2.f * d2) / 64.f);
    float ang = (float)s * inv;
    float cs = cosf(ang), sn = sinf(ang);

    long inb  = (long)tk * (NHEADS * ROPED) + h * ROPED;   // [tok, h*64]
    long outb = (long)tk * HDIM + h * HEADD;               // [tok, h*128]

    // roped halves
    float kx1 = (float)kr[inb + d2], kx2 = (float)kr[inb + d2 + 32];
    kout[outb + 64 + d2]      = (_Float16)(kx1 * cs - kx2 * sn);
    kout[outb + 64 + d2 + 32] = (_Float16)(kx2 * cs + kx1 * sn);
    float qx1 = (float)qr[inb + d2], qx2 = (float)qr[inb + d2 + 32];
    qout[outb + 64 + d2]      = (_Float16)((qx1 * cs - qx2 * sn) * QSCALE);
    qout[outb + 64 + d2 + 32] = (_Float16)((qx2 * cs + qx1 * sn) * QSCALE);

    // compressed halves (pass-through; q scaled)
    kout[outb + d2]      = kp[inb + d2];
    kout[outb + d2 + 32] = kp[inb + d2 + 32];
    qout[outb + d2]      = (_Float16)((float)qp[inb + d2] * QSCALE);
    qout[outb + d2 + 32] = (_Float16)((float)qp[inb + d2 + 32] * QSCALE);
  }
}

// =====================================================================
// Attention scores: one wave computes one 16x16 tile of S = q @ k^T for
// head h of batch b (scale already folded into q). Causal upper tiles skipped.
// q,k: [BATCH*SEQ, NH*128] f16.  scores: [NH, SEQ, SEQ] f32 (per-batch buffer).
// =====================================================================
__global__ __launch_bounds__(32) void attn_scores_kernel(const _Float16* __restrict__ q,
                                                         const _Float16* __restrict__ k,
                                                         float* __restrict__ scores,
                                                         int b) {
  int jt = blockIdx.x, it = blockIdx.y, h = blockIdx.z;
  if (jt > it) return;                       // causal: whole tile masked
  int lane = threadIdx.x;
  int laneRow = lane & 15, laneHi = lane >> 4;
  const _Float16* qb = q + (long)b * SEQ * HDIM + (long)h * HEADD;
  const _Float16* kb = k + (long)b * SEQ * HDIM + (long)h * HEADD;
  int i0 = it * 16, j0 = jt * 16;
  v8f c = {};
#pragma unroll
  for (int kk0 = 0; kk0 < HEADD; kk0 += 32) {
    v16h a, bf;
#pragma unroll
    for (int i = 0; i < 8; ++i) {
      int kl = ((i >> 2) << 4) + (laneHi << 3) + ((i & 3) << 1);
      a[2 * i]     = qb[(long)(i0 + laneRow) * HDIM + kk0 + kl];
      a[2 * i + 1] = qb[(long)(i0 + laneRow) * HDIM + kk0 + kl + 1];
    }
#pragma unroll
    for (int j = 0; j < 8; ++j) {
      int kl = ((j >> 2) << 4) + (laneHi << 3) + ((j & 3) << 1);
      // B[kk][n] = K[j0+n][kk]  (k^T)
      bf[2 * j]     = kb[(long)(j0 + laneRow) * HDIM + kk0 + kl];
      bf[2 * j + 1] = kb[(long)(j0 + laneRow) * HDIM + kk0 + kl + 1];
    }
    c = __builtin_amdgcn_wmma_f32_16x16x32_f16(false, a, false, bf, (short)0, c, false, false);
  }
  float* sc = scores + (long)h * SEQ * SEQ;
#pragma unroll
  for (int r = 0; r < 8; ++r)
    sc[(long)(i0 + (laneHi << 3) + r) * SEQ + j0 + laneRow] = c[r];
}

// =====================================================================
// Causal softmax over one score row -> f16 probs (zeros past the diagonal).
// grid = (SEQ, NH)
// =====================================================================
__global__ __launch_bounds__(256) void softmax_kernel(const float* __restrict__ scores,
                                                      _Float16* __restrict__ probs) {
  int i = blockIdx.x, h = blockIdx.y;
  const float* row = scores + (long)h * SEQ * SEQ + (long)i * SEQ;
  _Float16* prow   = probs  + (long)h * SEQ * SEQ + (long)i * SEQ;
  int L = i + 1;
  __shared__ float red[256];
  int tid = threadIdx.x;

  float mx = -1e30f;
  for (int j = tid; j < L; j += 256) mx = fmaxf(mx, row[j]);
  red[tid] = mx; __syncthreads();
  for (int st = 128; st > 0; st >>= 1) {
    if (tid < st) red[tid] = fmaxf(red[tid], red[tid + st]);
    __syncthreads();
  }
  mx = red[0]; __syncthreads();

  float s = 0.f;
  for (int j = tid; j < L; j += 256) s += expf(row[j] - mx);
  red[tid] = s; __syncthreads();
  for (int st = 128; st > 0; st >>= 1) {
    if (tid < st) red[tid] += red[tid + st];
    __syncthreads();
  }
  float inv = 1.f / red[0];

  for (int j = tid; j < SEQ; j += 256)
    prow[j] = (j < L) ? (_Float16)(expf(row[j] - mx) * inv) : (_Float16)0.f;
}

// =====================================================================
// silu-gate:  act = silu(g) * u / 8  (f32 in, f16 out)
// =====================================================================
__global__ __launch_bounds__(256) void silu_mul_kernel(const float* __restrict__ g,
                                                       const float* __restrict__ u,
                                                       _Float16* __restrict__ act, long n) {
  long i = (long)blockIdx.x * blockDim.x + threadIdx.x;
  long stride = (long)gridDim.x * blockDim.x;
  for (; i < n; i += stride) {
    float gv = g[i];
    float sv = gv / (1.f + expf(-gv));
    act[i] = (_Float16)(sv * u[i] * INV_SCALING);
  }
}

// =====================================================================
// Router: sigmoid(h_n @ W + b), top-2 of 7, normalized -> wout[tok][8]
// =====================================================================
__global__ __launch_bounds__(256) void router_kernel(const _Float16* __restrict__ hn,
                                                     const float* __restrict__ rw,
                                                     const float* __restrict__ rb,
                                                     float* __restrict__ wout) {
  int tk = blockIdx.x;
  __shared__ float red[256];
  __shared__ float logit[NREXP];
  float l[NREXP];
#pragma unroll
  for (int e = 0; e < NREXP; ++e) l[e] = 0.f;
  for (int kx = threadIdx.x; kx < HDIM; kx += 256) {
    float xv = (float)hn[(long)tk * HDIM + kx];
#pragma unroll
    for (int e = 0; e < NREXP; ++e) l[e] += xv * rw[kx * NREXP + e];
  }
  for (int e = 0; e < NREXP; ++e) {
    red[threadIdx.x] = l[e]; __syncthreads();
    for (int st = 128; st > 0; st >>= 1) {
      if (threadIdx.x < st) red[threadIdx.x] += red[threadIdx.x + st];
      __syncthreads();
    }
    if (threadIdx.x == 0) logit[e] = red[0] + rb[e];
    __syncthreads();
  }
  if (threadIdx.x == 0) {
    float p[NREXP];
#pragma unroll
    for (int e = 0; e < NREXP; ++e) p[e] = 1.f / (1.f + expf(-logit[e]));
    int i1 = 0;
    for (int e = 1; e < NREXP; ++e) if (p[e] > p[i1]) i1 = e;       // first max
    int i2 = (i1 == 0) ? 1 : 0;
    for (int e = 0; e < NREXP; ++e) if (e != i1 && p[e] > p[i2]) i2 = e;
    float s = p[i1] + p[i2];
    for (int e = 0; e < 8; ++e) wout[(long)tk * 8 + e] = 0.f;
    wout[(long)tk * 8 + i1] = p[i1] / s;
    wout[(long)tk * 8 + i2] = p[i2] / s;
  }
}

// =====================================================================
// out = h + shared_down          (element-wise, f32)
// =====================================================================
__global__ __launch_bounds__(256) void out_init_kernel(const float* __restrict__ h,
                                                       const float* __restrict__ d,
                                                       float* __restrict__ out, long n) {
  long i = (long)blockIdx.x * blockDim.x + threadIdx.x;
  long stride = (long)gridDim.x * blockDim.x;
  for (; i < n; i += stride) out[i] = h[i] + d[i];
}

// out += w[tok][e] * down        (element-wise, f32)
__global__ __launch_bounds__(256) void expert_accum_kernel(float* __restrict__ out,
                                                           const float* __restrict__ down,
                                                           const float* __restrict__ wr,
                                                           int e, long n) {
  long i = (long)blockIdx.x * blockDim.x + threadIdx.x;
  long stride = (long)gridDim.x * blockDim.x;
  for (; i < n; i += stride) {
    long tk = i >> 11;  // / HDIM(2048)
    out[i] += wr[tk * 8 + e] * down[i];
  }
}

// =====================================================================
// host orchestration
// =====================================================================
static inline int cdiv_h(long a, long b) { return (int)((a + b - 1) / b); }

extern "C" void kernel_launch(void* const* d_in, const int* in_sizes, int n_in,
                              void* d_out, int out_size, void* d_ws, size_t ws_size,
                              hipStream_t stream) {
  (void)in_sizes; (void)n_in; (void)out_size; (void)ws_size;
  const float* x        = (const float*)d_in[0];
  const float* ln1_w    = (const float*)d_in[1];
  const float* ln2_w    = (const float*)d_in[2];
  const float* kv_d_w   = (const float*)d_in[3];
  const float* q_d_w    = (const float*)d_in[4];
  const float* k_u_w    = (const float*)d_in[5];
  const float* v_u_w    = (const float*)d_in[6];
  const float* q_u_w    = (const float*)d_in[7];
  const float* rope_k_w = (const float*)d_in[8];
  const float* rope_q_w = (const float*)d_in[9];
  const float* o_w      = (const float*)d_in[10];
  const float* sh_gate  = (const float*)d_in[11];
  const float* sh_up    = (const float*)d_in[12];
  const float* sh_down  = (const float*)d_in[13];
  const float* r_gate   = (const float*)d_in[14];
  const float* r_up     = (const float*)d_in[15];
  const float* r_down   = (const float*)d_in[16];
  const float* router_w = (const float*)d_in[17];
  const float* router_b = (const float*)d_in[18];
  float* out = (float*)d_out;

  // -------- workspace bump allocator --------
  char* ws = (char*)d_ws;
  size_t off = 0;
  auto alloc = [&](size_t bytes) -> void* {
    off = (off + 255) & ~(size_t)255;
    void* p = ws + off;
    off += bytes;
    return p;
  };
  auto allocH = [&](long n) { return (_Float16*)alloc((size_t)n * 2); };
  auto allocF = [&](long n) { return (float*)alloc((size_t)n * 4); };

  // f16 weights
  _Float16* w_kvd = allocH((long)HDIM * LATD);
  _Float16* w_qd  = allocH((long)HDIM * LATD);
  _Float16* w_ku  = allocH((long)LATD * 1024);
  _Float16* w_vu  = allocH((long)LATD * HDIM);
  _Float16* w_qu  = allocH((long)LATD * 1024);
  _Float16* w_rk  = allocH((long)HDIM * 1024);
  _Float16* w_rq  = allocH((long)LATD * 1024);
  _Float16* w_o   = allocH((long)HDIM * HDIM);
  _Float16* w_sg  = allocH((long)HDIM * IDIM);
  _Float16* w_su  = allocH((long)HDIM * IDIM);
  _Float16* w_sd  = allocH((long)IDIM * HDIM);
  _Float16* w_rg  = allocH((long)NREXP * HDIM * IDIM);
  _Float16* w_ru  = allocH((long)NREXP * HDIM * IDIM);
  _Float16* w_rd  = allocH((long)NREXP * IDIM * HDIM);

  // activations
  _Float16* xn   = allocH((long)TOKENS * HDIM);
  _Float16* kvd  = allocH((long)TOKENS * LATD);
  _Float16* qd   = allocH((long)TOKENS * LATD);
  _Float16* kp   = allocH((long)TOKENS * 1024);
  _Float16* qp   = allocH((long)TOKENS * 1024);
  _Float16* kr   = allocH((long)TOKENS * 1024);
  _Float16* qr   = allocH((long)TOKENS * 1024);
  _Float16* vh   = allocH((long)TOKENS * HDIM);
  _Float16* kh   = allocH((long)TOKENS * HDIM);
  _Float16* qh   = allocH((long)TOKENS * HDIM);
  float*    scores = allocF((long)NHEADS * SEQ * SEQ);   // per-batch reuse
  _Float16* probs  = allocH((long)NHEADS * SEQ * SEQ);   // per-batch reuse
  _Float16* attnh  = allocH((long)TOKENS * HDIM);
  float*    proj   = allocF((long)TOKENS * HDIM);
  float*    hbuf   = allocF((long)TOKENS * HDIM);
  _Float16* hn     = allocH((long)TOKENS * HDIM);
  float*    gbuf   = allocF((long)TOKENS * IDIM);
  float*    ubuf   = allocF((long)TOKENS * IDIM);
  _Float16* acth   = allocH((long)TOKENS * IDIM);
  float*    dbuf   = allocF((long)TOKENS * HDIM);
  float*    wrt    = allocF((long)TOKENS * 8);

  auto cvt = [&](const float* s, _Float16* d, long n) {
    int g = cdiv_h(n, 256); if (g > 4096) g = 4096;
    hipLaunchKernelGGL(cvt_f32_f16, dim3(g), dim3(256), 0, stream, s, d, n);
  };
  cvt(kv_d_w,   w_kvd, (long)HDIM * LATD);
  cvt(q_d_w,    w_qd,  (long)HDIM * LATD);
  cvt(k_u_w,    w_ku,  (long)LATD * 1024);
  cvt(v_u_w,    w_vu,  (long)LATD * HDIM);
  cvt(q_u_w,    w_qu,  (long)LATD * 1024);
  cvt(rope_k_w, w_rk,  (long)HDIM * 1024);
  cvt(rope_q_w, w_rq,  (long)LATD * 1024);
  cvt(o_w,      w_o,   (long)HDIM * HDIM);
  cvt(sh_gate,  w_sg,  (long)HDIM * IDIM);
  cvt(sh_up,    w_su,  (long)HDIM * IDIM);
  cvt(sh_down,  w_sd,  (long)IDIM * HDIM);
  cvt(r_gate,   w_rg,  (long)NREXP * HDIM * IDIM);
  cvt(r_up,     w_ru,  (long)NREXP * HDIM * IDIM);
  cvt(r_down,   w_rd,  (long)NREXP * IDIM * HDIM);

  auto gemmH = [&](const _Float16* A, const _Float16* B, _Float16* C,
                   int M, int N, int K, int lda, int ldb, int ldc,
                   int bz, long sA, long sB, long sC) {
    hipLaunchKernelGGL(gemm_f16<true>, dim3(N / 128, M / 64, bz), dim3(256), 0, stream,
                       A, B, (void*)C, M, N, K, lda, ldb, ldc, sA, sB, sC);
  };
  auto gemmF = [&](const _Float16* A, const _Float16* B, float* C,
                   int M, int N, int K, int lda, int ldb, int ldc) {
    hipLaunchKernelGGL(gemm_f16<false>, dim3(N / 128, M / 64, 1), dim3(256), 0, stream,
                       A, B, (void*)C, M, N, K, lda, ldb, ldc, 0L, 0L, 0L);
  };

  // ---------------- attention ----------------
  hipLaunchKernelGGL(rmsnorm_kernel, dim3(TOKENS), dim3(256), 0, stream, x, ln1_w, xn);

  gemmH(xn,  w_kvd, kvd, TOKENS, LATD, HDIM, HDIM, LATD, LATD, 1, 0, 0, 0);   // kv_d
  gemmH(xn,  w_qd,  qd,  TOKENS, LATD, HDIM, HDIM, LATD, LATD, 1, 0, 0, 0);   // q_d
  gemmH(kvd, w_ku,  kp,  TOKENS, 1024, LATD, LATD, 1024, 1024, 1, 0, 0, 0);   // k_p
  gemmH(qd,  w_qu,  qp,  TOKENS, 1024, LATD, LATD, 1024, 1024, 1, 0, 0, 0);   // q_p
  gemmH(kvd, w_vu,  vh,  TOKENS, HDIM, LATD, LATD, HDIM, HDIM, 1, 0, 0, 0);   // v
  gemmH(xn,  w_rk,  kr,  TOKENS, 1024, HDIM, HDIM, 1024, 1024, 1, 0, 0, 0);   // k_r
  gemmH(qd,  w_rq,  qr,  TOKENS, 1024, LATD, LATD, 1024, 1024, 1, 0, 0, 0);   // q_r

  hipLaunchKernelGGL(rope_pack_kernel, dim3(TOKENS), dim3(256), 0, stream,
                     kp, qp, kr, qr, kh, qh);

  for (int b = 0; b < BATCH; ++b) {
    hipLaunchKernelGGL(attn_scores_kernel, dim3(SEQ / 16, SEQ / 16, NHEADS), dim3(32), 0,
                       stream, qh, kh, scores, b);
    hipLaunchKernelGGL(softmax_kernel, dim3(SEQ, NHEADS), dim3(256), 0, stream,
                       scores, probs);
    // attn_out[b,:,h*128:...] = probs[h] @ v[b,:,h]   (batched over heads)
    gemmH(probs, vh + (long)b * SEQ * HDIM, attnh + (long)b * SEQ * HDIM,
          SEQ, HEADD, SEQ, SEQ, HDIM, HDIM,
          NHEADS, (long)SEQ * SEQ, HEADD, HEADD);
  }

  gemmF(attnh, w_o, proj, TOKENS, HDIM, HDIM, HDIM, HDIM, HDIM);              // @ o_w
  hipLaunchKernelGGL(add_rmsnorm_kernel, dim3(TOKENS), dim3(256), 0, stream,
                     x, proj, ln2_w, hbuf, hn);                               // h, hn

  // ---------------- MoE ----------------
  const long NE = (long)TOKENS * HDIM;
  const long NI = (long)TOKENS * IDIM;

  // shared expert
  gemmF(hn, w_sg, gbuf, TOKENS, IDIM, HDIM, HDIM, IDIM, IDIM);
  gemmF(hn, w_su, ubuf, TOKENS, IDIM, HDIM, HDIM, IDIM, IDIM);
  hipLaunchKernelGGL(silu_mul_kernel, dim3(4096), dim3(256), 0, stream, gbuf, ubuf, acth, NI);
  gemmF(acth, w_sd, dbuf, TOKENS, HDIM, IDIM, IDIM, HDIM, HDIM);
  hipLaunchKernelGGL(out_init_kernel, dim3(4096), dim3(256), 0, stream, hbuf, dbuf, out, NE);

  // router
  hipLaunchKernelGGL(router_kernel, dim3(TOKENS), dim3(256), 0, stream,
                     hn, router_w, router_b, wrt);

  // routed experts (dense, weighted accumulate — matches reference)
  for (int e = 0; e < NREXP; ++e) {
    gemmF(hn, w_rg + (long)e * HDIM * IDIM, gbuf, TOKENS, IDIM, HDIM, HDIM, IDIM, IDIM);
    gemmF(hn, w_ru + (long)e * HDIM * IDIM, ubuf, TOKENS, IDIM, HDIM, HDIM, IDIM, IDIM);
    hipLaunchKernelGGL(silu_mul_kernel, dim3(4096), dim3(256), 0, stream, gbuf, ubuf, acth, NI);
    gemmF(acth, w_rd + (long)e * IDIM * HDIM, dbuf, TOKENS, HDIM, IDIM, IDIM, HDIM, HDIM);
    hipLaunchKernelGGL(expert_accum_kernel, dim3(4096), dim3(256), 0, stream,
                       out, dbuf, wrt, e, NE);
  }
}
